// VMamba_Feature_Extractor_4990751998230
// MI455X (gfx1250) — matrix operand order repaired and checked by reading, not compiled, not run on hardware
//
#include <hip/hip_runtime.h>
#include <math.h>

// ---------------------------------------------------------------------------
// CDNA5 (gfx1250) VMamba feature extractor.
// Channel GEMMs use v_wmma_f32_16x16x32_bf16 (bf16 A/B, f32 accumulate),
// fed from pre-swizzled LDS fragments (packed b32 stores, b128 loads).
// ---------------------------------------------------------------------------

typedef __attribute__((ext_vector_type(16))) __bf16        v16bf;
typedef __attribute__((ext_vector_type(8)))  float         v8f;
typedef __attribute__((ext_vector_type(8)))  unsigned int  v8u;

__device__ __forceinline__ unsigned short f2bf(float f) {
  unsigned int u = __builtin_bit_cast(unsigned int, f);
  u += 0x7fffu + ((u >> 16) & 1u);            // round-to-nearest-even
  return (unsigned short)(u >> 16);
}
__device__ __forceinline__ unsigned int pack2bf(float lo, float hi) {
  return (unsigned int)f2bf(lo) | ((unsigned int)f2bf(hi) << 16);
}

__device__ __forceinline__ float gelu_f(float x) {
  return 0.5f * x * (1.0f + erff(x * 0.70710678118654752f));
}

__device__ __forceinline__ float act_apply(float v, int act) {
  if (act == 1) return gelu_f(v);                    // exact GELU
  if (act == 2) return v / (1.0f + expf(-v));        // SiLU
  return v;
}

// ---------------------------------------------------------------------------
// GEMM: Y[M,N] = act(A[M,K] @ X[K,N] + bias) + res   (per batch slice)
// A row-major (weights), X channel-major (C x L activations).
// Block = 128 threads = 4 waves. Block tile M=16 x N=128; each wave owns a
// 16x32 slab (two 16x16 WMMA accumulators). K stepped by 32.
//
// LDS holds operands in this kernel's per-lane fragment order (contiguous k, same for A and B):
//   A 16x32 bf16: lane = m + 16*(k>>4), slot = k & 15
//   B 32x16 bf16: lane = (n&15) + 16*(k>>4), slot = k & 15
// Even/odd k land in adjacent slots of the same lane-row -> one packed b32
// store per f32 pair; each fragment read is 32 contiguous bytes per lane.
// Interior tiles take an unguarded float2 fast path (block-uniform branch).
// ---------------------------------------------------------------------------
__global__ __launch_bounds__(128)
void gemm_bf16_wmma_kernel(const float* __restrict__ A, const float* __restrict__ X,
                           const float* __restrict__ bias, const float* __restrict__ res,
                           float* __restrict__ Y, int M, int N, int K,
                           long long sA, long long sX, long long sRes, long long sY,
                           int act) {
  __shared__ __attribute__((aligned(32))) unsigned int AsF[32][8];      // 1 KB
  __shared__ __attribute__((aligned(32))) unsigned int XsF[8][32][8];   // 8 KB
  int bz = blockIdx.z;
  A += (long long)bz * sA;
  X += (long long)bz * sX;
  Y += (long long)bz * sY;
  if (res) res += (long long)bz * sRes;

  int m0  = blockIdx.y * 16;
  int n0  = blockIdx.x * 128;
  int tid = threadIdx.x;
  int lane = tid & 31, wv = tid >> 5;
  int half = lane >> 4, q = lane & 15;

  v8f acc0 = {}, acc1 = {};
  for (int k0 = 0; k0 < K; k0 += 32) {
    // block-uniform interior test (all our K are even; N odd only in stage 4)
    bool full = (m0 + 16 <= M) && (k0 + 32 <= K) && (n0 + 128 <= N) &&
                ((N & 1) == 0) && ((K & 1) == 0);
    if (full) {
      // ---- A tile fast path: unguarded float2 loads ----
#pragma unroll
      for (int p = tid; p < 256; p += 128) {
        int m = p >> 4;
        int k = (p & 15) * 2;
        float2 v = *(const float2*)&A[(long long)(m0 + m) * K + (k0 + k)];
        AsF[m + 16 * (k >> 4)][(k & 15) >> 1] =
            pack2bf(v.x, v.y);
      }
      // ---- X tile fast path: 2x2 quads (k-pair x n-pair) ----
#pragma unroll
      for (int p = tid; p < 1024; p += 128) {
        int n = (p & 63) * 2;
        int k = (p >> 6) * 2;
        const float* base = &X[(long long)(k0 + k) * N + (n0 + n)];
        float2 r0 = *(const float2*)base;          // row k,   cols n..n+1
        float2 r1 = *(const float2*)(base + N);    // row k+1, cols n..n+1
        int grp = n >> 4, row = (n & 15) + 16 * (k >> 4), slot = (k & 15) >> 1;
        XsF[grp][row + 0][slot] = pack2bf(r0.x, r1.x);
        XsF[grp][row + 1][slot] = pack2bf(r0.y, r1.y);
      }
    } else {
      // ---- guarded edge path ----
#pragma unroll
      for (int p = tid; p < 256; p += 128) {
        int m = p >> 4;
        int k = (p & 15) * 2;
        float v0 = 0.0f, v1 = 0.0f;
        if ((m0 + m) < M) {
          const float* ap = &A[(long long)(m0 + m) * K + (k0 + k)];
          if ((k0 + k)     < K) v0 = ap[0];
          if ((k0 + k + 1) < K) v1 = ap[1];
        }
        AsF[m + 16 * (k >> 4)][(k & 15) >> 1] =
            pack2bf(v0, v1);
      }
#pragma unroll
      for (int p = tid; p < 2048; p += 128) {
        int n = p & 127;
        int k = (p >> 7) * 2;
        float v0 = 0.0f, v1 = 0.0f;
        if ((n0 + n) < N) {
          if ((k0 + k)     < K) v0 = X[(long long)(k0 + k)     * N + (n0 + n)];
          if ((k0 + k + 1) < K) v1 = X[(long long)(k0 + k + 1) * N + (n0 + n)];
        }
        XsF[n >> 4][(n & 15) + 16 * (k >> 4)][(k & 15) >> 1] = pack2bf(v0, v1);
      }
    }
    // hint next K-tile of activations toward the caches
    {
      int pr = k0 + 32 + (tid >> 2);
      if (pr < K) __builtin_prefetch(&X[(long long)pr * N + n0], 0, 1);
    }
    __syncthreads();

    v16bf af = __builtin_bit_cast(v16bf, *(const v8u*)&AsF[lane][0]);
    v16bf b0 = __builtin_bit_cast(v16bf, *(const v8u*)&XsF[2 * wv + 0][lane][0]);
    v16bf b1 = __builtin_bit_cast(v16bf, *(const v8u*)&XsF[2 * wv + 1][lane][0]);
    acc0 = __builtin_amdgcn_wmma_f32_16x16x32_bf16(false, af, false, b0,
                                                   (short)0, acc0, false, false);
    acc1 = __builtin_amdgcn_wmma_f32_16x16x32_bf16(false, af, false, b1,
                                                   (short)0, acc1, false, false);
    __syncthreads();
  }

  // C/D layout: VGPR r -> M = r (lanes 0-15) / r+8 (lanes 16-31), N = lane&15
#pragma unroll
  for (int t = 0; t < 2; ++t) {
    int n = n0 + (2 * wv + t) * 16 + q;
#pragma unroll
    for (int r = 0; r < 8; ++r) {
      int m = m0 + r + 8 * half;
      if (m < M && n < N) {
        float v = (t == 0) ? acc0[r] : acc1[r];
        if (bias) v += bias[m];
        v = act_apply(v, act);
        if (res) v += res[(long long)m * N + n];
        Y[(long long)m * N + n] = v;
      }
    }
  }
}

// ---------------------------------------------------------------------------
// Generic 3x3 conv (pad=1), supports stride, groups (depthwise), and channel
// replication for the 1->3 concat feeding patch_embed conv1.
// ---------------------------------------------------------------------------
__global__ void conv3x3_kernel(const float* __restrict__ x, const float* __restrict__ wgt,
                               const float* __restrict__ bias, float* __restrict__ y,
                               int B, int Cin_data, int cin_pg, int Cout,
                               int Hin, int Win, int Hout, int Wout,
                               int stride, int groups, int act) {
  long long idx = (long long)blockIdx.x * blockDim.x + threadIdx.x;
  long long total = (long long)B * Cout * Hout * Wout;
  if (idx >= total) return;
  int wo = (int)(idx % Wout); long long t = idx / Wout;
  int ho = (int)(t % Hout); t /= Hout;
  int co = (int)(t % Cout); int n = (int)(t / Cout);
  int cpgo = Cout / groups;
  int g = co / cpgo;
  float acc = bias ? bias[co] : 0.0f;
  for (int i = 0; i < cin_pg; ++i) {
    int ci = (g * cin_pg + i) % Cin_data;   // replicated-channel trick for conv1
    const float* xp = x + ((long long)n * Cin_data + ci) * Hin * Win;
    const float* wp = wgt + ((long long)co * cin_pg + i) * 9;
#pragma unroll
    for (int kh = 0; kh < 3; ++kh) {
      int hi = ho * stride - 1 + kh;
      if (hi < 0 || hi >= Hin) continue;
#pragma unroll
      for (int kw = 0; kw < 3; ++kw) {
        int wi = wo * stride - 1 + kw;
        if (wi < 0 || wi >= Win) continue;
        acc = fmaf(xp[(long long)hi * Win + wi], wp[kh * 3 + kw], acc);
      }
    }
  }
  y[idx] = act_apply(acc, act);
}

// ---------------------------------------------------------------------------
// LayerNorm over channel dim per pixel (biased var), optional fused GELU.
// One thread owns one (b, pixel) column -> in-place safe.
// ---------------------------------------------------------------------------
__global__ void ln2d_kernel(const float* __restrict__ x, const float* __restrict__ g,
                            const float* __restrict__ b, float* __restrict__ y,
                            int C, long long HW, long long total, int act) {
  long long idx = (long long)blockIdx.x * blockDim.x + threadIdx.x;
  if (idx >= total) return;
  long long n = idx / HW, l = idx % HW;
  const float* xp = x + n * C * HW + l;
  float s = 0.0f, sq = 0.0f;
  for (int c = 0; c < C; ++c) { float v = xp[(long long)c * HW]; s += v; sq += v * v; }
  float mu = s / C;
  float var = sq / C - mu * mu;
  float rs = rsqrtf(var + 1e-5f);
  float* yp = y + n * C * HW + l;
  for (int c = 0; c < C; ++c) {
    float v = (xp[(long long)c * HW] - mu) * rs * g[c] + b[c];
    yp[(long long)c * HW] = act_apply(v, act);
  }
}

// (B,D,H,W) -> (B,4,D,L): row-major, col-major, and reverses
__global__ void cross_scan_kernel(const float* __restrict__ x, float* __restrict__ xs,
                                  int D, int H, int W, long long total) {
  long long idx = (long long)blockIdx.x * blockDim.x + threadIdx.x;
  if (idx >= total) return;
  int L = H * W;
  int l = (int)(idx % L); long long t = idx / L;
  int dd = (int)(t % D); t /= D;
  int k = (int)(t % 4); int b = (int)(t / 4);
  int le = (k >= 2) ? (L - 1 - l) : l;
  int src;
  if (k & 1) { int ww = le / H, hh = le % H; src = hh * W + ww; }
  else       { src = le; }
  xs[idx] = x[((long long)b * D + dd) * L + src];
}

// inverse of cross_scan, summing the 4 direction outputs -> (B,D,L)
__global__ void cross_merge_kernel(const float* __restrict__ ys, float* __restrict__ y,
                                   int D, int H, int W, long long total) {
  long long idx = (long long)blockIdx.x * blockDim.x + threadIdx.x;
  if (idx >= total) return;
  int L = H * W;
  int l = (int)(idx % L); long long t = idx / L;
  int dd = (int)(t % D); int b = (int)(t / D);
  const float* y0 = ys + ((long long)(b * 4 + 0) * D + dd) * L;
  const float* y1 = ys + ((long long)(b * 4 + 1) * D + dd) * L;
  const float* y2 = ys + ((long long)(b * 4 + 2) * D + dd) * L;
  const float* y3 = ys + ((long long)(b * 4 + 3) * D + dd) * L;
  int hh = l / W, ww = l % W;
  int lt = ww * H + hh;
  y[idx] = y0[l] + y2[L - 1 - l] + y1[lt] + y3[L - 1 - lt];
}

// selective scan, d_state == 1: per-channel scalar linear recurrence over L
__global__ void selscan_kernel(const float* __restrict__ u, const float* __restrict__ dt,
                               const float* __restrict__ xdbl,
                               const float* __restrict__ A_logs, const float* __restrict__ Ds,
                               const float* __restrict__ dtb, float* __restrict__ yo,
                               int D, int L, int R2, int total) {
  int idx = blockIdx.x * blockDim.x + threadIdx.x;
  if (idx >= total) return;
  int dd = idx % D; int k = (idx / D) % 4; int b = idx / (4 * D);
  int kd = k * D + dd;
  float Aa = -expf(A_logs[kd]);
  float Dk = Ds[kd];
  float db = dtb[kd];
  long long chan = ((long long)(b * 4 + k) * D + dd) * (long long)L;
  const float* up = u + chan;
  const float* dp = dt + chan;
  const float* Bp = xdbl + ((long long)(b * 4 + k) * R2 + (R2 - 2)) * (long long)L;
  const float* Cp = Bp + L;
  float* yp = yo + chan;
  float h = 0.0f;
  for (int l = 0; l < L; ++l) {
    float uv = up[l];
    float dl = dp[l] + db;
    dl = (dl > 20.0f) ? dl : log1pf(expf(dl));   // softplus
    float a = expf(dl * Aa);
    h = a * h + dl * Bp[l] * uv;
    yp[l] = h * Cp[l] + uv * Dk;
  }
}

__global__ void copy_kernel(const float* __restrict__ s, float* __restrict__ d, long long n) {
  long long i = (long long)blockIdx.x * blockDim.x + threadIdx.x;
  if (i < n) d[i] = s[i];
}

// ---------------------------------------------------------------------------
// Host orchestration
// ---------------------------------------------------------------------------
extern "C" void kernel_launch(void* const* d_in, const int* in_sizes, int n_in,
                              void* d_out, int out_size, void* d_ws, size_t ws_size,
                              hipStream_t stream) {
  (void)in_sizes; (void)n_in; (void)out_size; (void)ws_size;
  const int B = 2;

  // ---- unpack params: JAX tree order (dict keys sorted, lists in order) ----
  int idx = 0;
  const float* PE[8];                         // b1,b2,be1,be2,g1,g2,w1,w2
  for (int i = 0; i < 8; ++i) PE[i] = (const float*)d_in[idx++];
  static const int depths[4] = {2, 2, 8, 2};
  // block tensor order (sorted keys):
  // 0 A_logs 1 Ds 2 dt_b 3 dt_w 4 dwconv_w 5 fc1_b 6 fc1_w 7 fc2_b 8 fc2_w
  // 9 in_w 10 n1b 11 n1g 12 n2b 13 n2g 14 out_norm_b 15 out_norm_g 16 out_w 17 x_proj_w
  const float* BLK[4][8][18];
  const float* DWN[3][4];                     // b, bb, g, w
  for (int s = 0; s < 4; ++s) {
    for (int bl = 0; bl < depths[s]; ++bl)
      for (int t = 0; t < 18; ++t) BLK[s][bl][t] = (const float*)d_in[idx++];
    if (s < 3)
      for (int t = 0; t < 4; ++t) DWN[s][t] = (const float*)d_in[idx++];
  }
  const float* x_in = (const float*)d_in[idx++];   // index 272
  float* out = (float*)d_out;

  // ---- workspace carve (floats) ----
  float* wsp = (float*)d_ws;
  size_t off = 0;
  auto carve = [&](size_t n) -> float* { float* p = wsp + off; off += n; return p; };
  float* big0 = carve(1204224);   // (2,48,112,112)
  float* bufA = carve(602112);    // current activations (B,d,L)
  float* bufB = carve(602112);
  float* bufC = carve(602112);
  float* bufD = carve(602112);
  float* ymrg = carve(602112);
  float* xsB  = carve(2408448);   // (B,4,D,L)
  float* dtsB = carve(2408448);
  float* ysB  = carve(2408448);
  float* hid  = carve(2408448);   // (B,4d,L)
  float* xdbl = carve(200704);    // (B,4,R+2,L)

  auto g1d = [](long long n) { return dim3((unsigned)((n + 255) / 256)); };

  auto conv = [&](const float* x, const float* wg, const float* bi, float* y,
                  int Cind, int cpg, int Co, int Hi, int Wi, int Ho, int Wo,
                  int st, int gr, int act) {
    long long tot = (long long)B * Co * Ho * Wo;
    conv3x3_kernel<<<g1d(tot), 256, 0, stream>>>(x, wg, bi, y, B, Cind, cpg, Co,
                                                 Hi, Wi, Ho, Wo, st, gr, act);
  };
  auto ln = [&](const float* x, const float* g, const float* b, float* y,
                int C, long long HW, int act) {
    long long tot = (long long)B * HW;
    ln2d_kernel<<<g1d(tot), 256, 0, stream>>>(x, g, b, y, C, HW, tot, act);
  };
  auto gemm = [&](const float* A, const float* X, const float* bi, const float* rs,
                  float* Y, int M, int N, int K, int batch,
                  long long sA, long long sX, long long sR, long long sY, int act) {
    dim3 grid((unsigned)((N + 127) / 128), (unsigned)((M + 15) / 16), (unsigned)batch);
    gemm_bf16_wmma_kernel<<<grid, 128, 0, stream>>>(A, X, bi, rs, Y, M, N, K,
                                                    sA, sX, sR, sY, act);
  };

  // ---- patch embed ----
  conv(x_in, PE[6], PE[0], big0, 1, 3, 48, 224, 224, 112, 112, 2, 1, 0);
  ln(big0, PE[4], PE[2], big0, 48, 112LL * 112, 1);              // LN + GELU
  conv(big0, PE[7], PE[1], bufA, 48, 48, 96, 112, 112, 56, 56, 2, 1, 0);
  ln(bufA, PE[5], PE[3], bufA, 96, 3136, 0);

  // ---- stages ----
  static const int dims[4] = {96, 192, 384, 768};
  static const int Hs[4]   = {56, 28, 14, 7};
  size_t featOff = 0;
  for (int s = 0; s < 4; ++s) {
    int d = dims[s], H = Hs[s], W = H, L = H * W;
    int R = (d + 15) / 16, R2 = R + 2;
    long long dL = (long long)d * L;
    for (int bl = 0; bl < depths[s]; ++bl) {
      const float* const* P = BLK[s][bl];
      // ---- SS2D branch ----
      ln(bufA, P[11], P[10], bufB, d, L, 0);                               // n1
      gemm(P[9], bufB, nullptr, nullptr, bufC, d, L, d, B, 0, dL, 0, dL, 0);   // in_proj
      conv(bufC, P[4], nullptr, bufD, d, 1, d, H, W, H, W, 1, d, 2);       // dwconv+SiLU
      {
        long long tot = (long long)B * 4 * d * L;
        cross_scan_kernel<<<g1d(tot), 256, 0, stream>>>(bufD, xsB, d, H, W, tot);
      }
      for (int k = 0; k < 4; ++k)                                          // x_proj
        gemm(P[17] + (size_t)k * R2 * d, xsB + (size_t)k * dL, nullptr, nullptr,
             xdbl + (size_t)k * R2 * L, R2, L, d, B,
             0, 4 * dL, 0, (long long)4 * R2 * L, 0);
      for (int k = 0; k < 4; ++k)                                          // dt proj
        gemm(P[3] + (size_t)k * d * R, xdbl + (size_t)k * R2 * L, nullptr, nullptr,
             dtsB + (size_t)k * dL, d, L, R, B,
             0, (long long)4 * R2 * L, 0, 4 * dL, 0);
      {
        int tot = B * 4 * d;
        selscan_kernel<<<g1d(tot), 256, 0, stream>>>(xsB, dtsB, xdbl,
                                                     P[0], P[1], P[2], ysB,
                                                     d, L, R2, tot);
      }
      {
        long long tot = (long long)B * d * L;
        cross_merge_kernel<<<g1d(tot), 256, 0, stream>>>(ysB, ymrg, d, H, W, tot);
      }
      ln(ymrg, P[15], P[14], bufB, d, L, 0);                               // out_norm
      gemm(P[16], bufB, nullptr, bufA, bufA, d, L, d, B, 0, dL, dL, dL, 0);    // out_proj + residual
      // ---- MLP branch ----
      ln(bufA, P[13], P[12], bufB, d, L, 0);                               // n2
      gemm(P[6], bufB, P[5], nullptr, hid, 4 * d, L, d, B, 0, dL, 0, 4 * dL, 1); // fc1 + GELU
      gemm(P[8], hid, P[7], bufA, bufA, d, L, 4 * d, B, 0, 4 * dL, dL, dL, 0);   // fc2 + residual
    }
    // emit stage feature
    {
      long long n = (long long)B * d * L;
      copy_kernel<<<g1d(n), 256, 0, stream>>>(bufA, out + featOff, n);
      featOff += (size_t)n;
    }
    // downsample
    if (s < 3) {
      int H2 = H / 2;
      conv(bufA, DWN[s][3], DWN[s][0], bufC, d, d, 2 * d, H, W, H2, H2, 2, 1, 0);
      ln(bufC, DWN[s][2], DWN[s][1], bufC, 2 * d, (long long)H2 * H2, 0);
      long long n = (long long)B * 2 * d * H2 * H2;
      copy_kernel<<<g1d(n), 256, 0, stream>>>(bufC, bufA, n);
    }
  }
}
